// DartAwareBlock_46815143526788
// MI455X (gfx1250) — matrix operand order repaired and checked by reading, not compiled, not run on hardware
//
#include <hip/hip_runtime.h>
#include <hip/hip_bf16.h>

typedef __bf16 bf16_t;
typedef __attribute__((ext_vector_type(16))) __bf16 v16bf;
typedef __attribute__((ext_vector_type(8)))  __bf16 v8bf;
typedef __attribute__((ext_vector_type(8)))  float  v8f;

#define B_   4
#define N_   256
#define HID_ 128
#define H_   8
#define DH_  16
#define E_   32
#define G_   64

// ---------------------------------------------------------------------------
// WMMA fragment helpers (v_wmma_f32_16x16x32_bf16, wave32)
// A 16x32 bf16: lane m = lane&15; lo lanes hold K {0..7,16..23}, hi {8..15,24..31}
// B 32x16 bf16: lane n = lane&15; lo lanes hold K {0..7,16..23}, hi {8..15,24..31}
// C/D 16x16 f32: lane col n = lane&15; VGPR g holds row m = g + 8*(lane>>4)
// ---------------------------------------------------------------------------
__device__ __forceinline__ v16bf load_a_frag(const bf16_t* row_base, int lane) {
  const int koff = (lane >> 4) * 8;
  union { v16bf v; v8bf h[2]; } u;
  u.h[0] = *(const v8bf*)(row_base + koff);        // K {koff..koff+7}
  u.h[1] = *(const v8bf*)(row_base + 16 + koff);   // K {16+koff..}
  return u.v;
}

// packed weight layout: wp[kb][n][kk]  (kb = K/32 block, n = out col, kk = 0..31)
__device__ __forceinline__ v16bf load_b_frag(const bf16_t* wp, int kb, int n, int lane) {
  const bf16_t* p = wp + ((size_t)kb * HID_ + n) * 32 + (lane >> 4) * 8;
  union { v16bf v; v8bf h[2]; } u;
  u.h[0] = *(const v8bf*)p;
  u.h[1] = *(const v8bf*)(p + 16);
  return u.v;
}

// ---------------------------------------------------------------------------
// pack f32 [K][128] weights -> bf16 fragment-native [K/32][128][32]
// ---------------------------------------------------------------------------
__global__ void pack_w_kernel(const float* __restrict__ src,
                              bf16_t* __restrict__ dst, int K) {
  const int idx = blockIdx.x * blockDim.x + threadIdx.x;
  if (idx >= K * HID_) return;
  const int k  = idx / HID_;
  const int n  = idx - k * HID_;
  const int kb = k >> 5, kk = k & 31;
  dst[((size_t)kb * HID_ + n) * 32 + kk] = (bf16_t)src[idx];
}

// ---------------------------------------------------------------------------
// Unified 2-layer MLP + residual + LayerNorm + ReLU, 32-row tiles, WMMA bf16.
// Each of 8 waves owns a 32x16 output slab (two 16x16 accumulators sharing
// each B fragment -> 2 WMMAs per weight-fragment load).
// mode 0: dart update (rows = B*N*N, K1 = 416, feats [src|dst|dart|edge])
// mode 1: node update (rows = B*N,   K1 = 448, feats [node|in_msg|out_msg|glob])
// ---------------------------------------------------------------------------
__global__ __launch_bounds__(256)
void mlp_ln_kernel(const float* __restrict__ f0, const float* __restrict__ f1,
                   const float* __restrict__ f2, const float* __restrict__ f3,
                   const bf16_t* __restrict__ w1, const float* __restrict__ b1,
                   const bf16_t* __restrict__ w2, const float* __restrict__ b2,
                   const float* __restrict__ ln_g, const float* __restrict__ ln_b,
                   const float* __restrict__ resid, float* __restrict__ out,
                   int K1, int mode) {
  __shared__ alignas(16) bf16_t sX[32 * 448];
  __shared__ alignas(16) bf16_t sH1[32 * HID_];
  __shared__ float sOut[32 * HID_];

  const int  tid  = threadIdx.x;
  const int  lane = tid & 31;
  const int  wave = tid >> 5;
  const long row0 = (long)blockIdx.x * 32;

  // ---- gather activations into LDS as bf16 ----
  if (mode == 0) {
    const int bi = (int)(row0 / N_);   // b*N + i (fixed per tile; 32 | N)
    const int j0 = (int)(row0 % N_);
    const int b  = bi / N_;
    const float* srcrow = f0 + (size_t)bi * HID_;
    for (int idx = tid; idx < 32 * 416; idx += 256) {
      const int r = idx / 416;
      const int f = idx - r * 416;
      const int j = j0 + r;
      float val;
      if (f < 128)      val = srcrow[f];
      else if (f < 256) val = f0[(size_t)(b * N_ + j) * HID_ + (f - 128)];
      else if (f < 384) val = f2[((size_t)bi * N_ + j) * HID_ + (f - 256)];
      else              val = f3[((size_t)bi * N_ + j) * E_   + (f - 384)];
      sX[r * 448 + f] = (bf16_t)val;
    }
  } else {
    for (int idx = tid; idx < 32 * 448; idx += 256) {
      const int  r   = idx / 448;
      const int  f   = idx - r * 448;
      const long row = row0 + r;           // b*N + i
      const int  b   = (int)(row / N_);
      float val;
      if (f < 128)      val = f0[(size_t)row * HID_ + f];
      else if (f < 256) val = f1[(size_t)row * HID_ + (f - 128)];
      else if (f < 384) val = f2[(size_t)row * HID_ + (f - 256)];
      else              val = f3[(size_t)b * G_ + (f - 384)];
      sX[r * 448 + f] = (bf16_t)val;
    }
  }
  __syncthreads();

  const int m   = lane & 15;
  const int hi  = lane >> 4;
  const int nc  = wave * 16 + (lane & 15);   // output feature column
  const int KB1 = K1 >> 5;

  // ---- layer 1: [32 x K1] @ [K1 x 128], ReLU ----
  v8f acc0 = {0.f, 0.f, 0.f, 0.f, 0.f, 0.f, 0.f, 0.f};
  v8f acc1 = acc0;
  for (int kb = 0; kb < KB1; ++kb) {
    const v16bf bm = load_b_frag(w1, kb, nc, lane);
    const v16bf a0 = load_a_frag(sX + m * 448 + kb * 32, lane);
    const v16bf a1 = load_a_frag(sX + (m + 16) * 448 + kb * 32, lane);
    acc0 = __builtin_amdgcn_wmma_f32_16x16x32_bf16(false, a0, false, bm,
                                                   (short)0, acc0, false, false);
    acc1 = __builtin_amdgcn_wmma_f32_16x16x32_bf16(false, a1, false, bm,
                                                   (short)0, acc1, false, false);
  }
  const float bias1 = b1[nc];
#pragma unroll
  for (int g = 0; g < 8; ++g) {
    const int mr = g + 8 * hi;
    sH1[mr * HID_ + nc]        = (bf16_t)fmaxf(acc0[g] + bias1, 0.f);
    sH1[(mr + 16) * HID_ + nc] = (bf16_t)fmaxf(acc1[g] + bias1, 0.f);
  }
  __syncthreads();

  // ---- layer 2: [32 x 128] @ [128 x 128] + residual ----
  v8f acc2 = {0.f, 0.f, 0.f, 0.f, 0.f, 0.f, 0.f, 0.f};
  v8f acc3 = acc2;
#pragma unroll
  for (int kb = 0; kb < 4; ++kb) {
    const v16bf bm = load_b_frag(w2, kb, nc, lane);
    const v16bf a0 = load_a_frag(sH1 + m * HID_ + kb * 32, lane);
    const v16bf a1 = load_a_frag(sH1 + (m + 16) * HID_ + kb * 32, lane);
    acc2 = __builtin_amdgcn_wmma_f32_16x16x32_bf16(false, a0, false, bm,
                                                   (short)0, acc2, false, false);
    acc3 = __builtin_amdgcn_wmma_f32_16x16x32_bf16(false, a1, false, bm,
                                                   (short)0, acc3, false, false);
  }
  const float bias2 = b2[nc];
#pragma unroll
  for (int g = 0; g < 8; ++g) {
    const int mr = g + 8 * hi;
    sOut[mr * HID_ + nc] =
        acc2[g] + bias2 + resid[(size_t)(row0 + mr) * HID_ + nc];
    sOut[(mr + 16) * HID_ + nc] =
        acc3[g] + bias2 + resid[(size_t)(row0 + mr + 16) * HID_ + nc];
  }
  __syncthreads();

  // ---- LayerNorm + ReLU (one row per thread, threads 0..31) ----
  if (tid < 32) {
    const float* xr = sOut + tid * HID_;
    float mean = 0.f;
    for (int f = 0; f < HID_; ++f) mean += xr[f];
    mean *= (1.f / HID_);
    float var = 0.f;
    for (int f = 0; f < HID_; ++f) { const float d = xr[f] - mean; var += d * d; }
    var *= (1.f / HID_);
    const float inv = rsqrtf(var + 1e-5f);
    float* orow = out + (size_t)(row0 + tid) * HID_;
    for (int f = 0; f < HID_; ++f) {
      const float y = (xr[f] - mean) * inv * ln_g[f] + ln_b[f];
      orow[f] = fmaxf(y, 0.f);
    }
  }
}

// ---------------------------------------------------------------------------
// Fused dart attention per (b, i). Algebraic folding:
//   scores[h,j] = dart_j . (kw_h @ q_h) + q_h . kb_h
//   out_h       = (sum_j attn[h,j] dart_j) @ vw_h + vb_h
// so k/v are never materialized (saves ~34 GFLOP + ~270 MB traffic).
// ---------------------------------------------------------------------------
__global__ __launch_bounds__(256)
void dart_attn_kernel(const float* __restrict__ node_h, const float* __restrict__ dart,
                      const float* __restrict__ mask,
                      const float* __restrict__ qw, const float* __restrict__ qb,
                      const float* __restrict__ kw, const float* __restrict__ kb,
                      const float* __restrict__ vw, const float* __restrict__ vb,
                      const float* __restrict__ ow, const float* __restrict__ ob,
                      float* __restrict__ msg, float* __restrict__ attn_out,
                      int transpose) {
  __shared__ float s_node[HID_];
  __shared__ float s_q[HID_];
  __shared__ float s_qk[HID_ * H_];
  __shared__ float s_qb[H_];
  __shared__ float s_sc[H_ * N_];
  __shared__ float s_red[H_ * 32];
  __shared__ float s_m[H_ * HID_];
  __shared__ float s_o[HID_];

  const int tid = threadIdx.x;
  const int b   = blockIdx.x / N_;
  const int i   = blockIdx.x % N_;

  if (tid < HID_) s_node[tid] = node_h[(size_t)(b * N_ + i) * HID_ + tid];
  __syncthreads();

  // q = node_row @ qw + qb
  if (tid < HID_) {
    float a = qb[tid];
    for (int k = 0; k < HID_; ++k) a += s_node[k] * qw[k * HID_ + tid];
    s_q[tid] = a;
  }
  __syncthreads();

  // fold q into kw: qk[dk][h] = sum_d kw[dk][h*16+d] * q[h*16+d]
  for (int idx = tid; idx < HID_ * H_; idx += 256) {
    const int dk = idx >> 3, h = idx & 7;
    float a = 0.f;
    for (int d = 0; d < DH_; ++d) a += kw[dk * HID_ + h * DH_ + d] * s_q[h * DH_ + d];
    s_qk[idx] = a;
  }
  if (tid < H_) {
    float a = 0.f;
    for (int d = 0; d < DH_; ++d) a += s_q[tid * DH_ + d] * kb[tid * DH_ + d];
    s_qb[tid] = a;
  }
  __syncthreads();

  // scores: one j per thread (blockDim == N_)
  {
    const int j = tid;
    const float* drow = transpose ? dart + ((size_t)(b * N_ + j) * N_ + i) * HID_
                                  : dart + ((size_t)(b * N_ + i) * N_ + j) * HID_;
    float acc[H_];
#pragma unroll
    for (int h = 0; h < H_; ++h) acc[h] = 0.f;
    for (int k = 0; k < HID_; ++k) {
      const float dv = drow[k];
      const float* qkr = s_qk + k * H_;
#pragma unroll
      for (int h = 0; h < H_; ++h) acc[h] += dv * qkr[h];
    }
    const float mval = transpose ? mask[(size_t)(b * N_ + j) * N_ + i]
                                 : mask[(size_t)(b * N_ + i) * N_ + j];
    const bool valid = (j == i) || (mval > 0.5f);
#pragma unroll
    for (int h = 0; h < H_; ++h) {
      float s = (acc[h] + s_qb[h]) * 0.25f;          // 1/sqrt(DH)
      s = (s >= 0.f) ? s : 0.2f * s;                 // LeakyReLU(0.2)
      s_sc[h * N_ + j] = valid ? s : -1e9f;
    }
  }
  __syncthreads();

  // softmax per head (32 threads per head), write attn to d_out
  {
    const int h = tid >> 5, lane = tid & 31;
    float mx = -3.4e38f;
    for (int t = 0; t < 8; ++t) mx = fmaxf(mx, s_sc[h * N_ + lane + 32 * t]);
    s_red[h * 32 + lane] = mx;
    __syncthreads();
    if (lane == 0) {
      float m2 = -3.4e38f;
      for (int l = 0; l < 32; ++l) m2 = fmaxf(m2, s_red[h * 32 + l]);
      s_red[h * 32] = m2;
    }
    __syncthreads();
    mx = s_red[h * 32];
    __syncthreads();
    float sum = 0.f;
    for (int t = 0; t < 8; ++t) {
      const float e = __expf(s_sc[h * N_ + lane + 32 * t] - mx);
      s_sc[h * N_ + lane + 32 * t] = e;
      sum += e;
    }
    s_red[h * 32 + lane] = sum;
    __syncthreads();
    if (lane == 0) {
      float s2 = 0.f;
      for (int l = 0; l < 32; ++l) s2 += s_red[h * 32 + l];
      s_red[h * 32] = s2;
    }
    __syncthreads();
    const float inv = 1.f / s_red[h * 32];
    float* arow = attn_out + ((size_t)(b * H_ + h) * N_ + i) * N_;
    for (int t = 0; t < 8; ++t) {
      const int j = lane + 32 * t;
      const float a = s_sc[h * N_ + j] * inv;
      s_sc[h * N_ + j] = a;
      arow[j] = a;
    }
  }
  __syncthreads();

  // m[h][d] = sum_j attn[h][j] * dart_j[d]
  {
    const int d  = tid & 127;
    const int hg = tid >> 7;   // 0..1 -> heads hg*4 .. hg*4+3
    float a4[4] = {0.f, 0.f, 0.f, 0.f};
    for (int j = 0; j < N_; ++j) {
      const float dv = transpose ? dart[((size_t)(b * N_ + j) * N_ + i) * HID_ + d]
                                 : dart[((size_t)(b * N_ + i) * N_ + j) * HID_ + d];
#pragma unroll
      for (int t = 0; t < 4; ++t) a4[t] += s_sc[(hg * 4 + t) * N_ + j] * dv;
    }
#pragma unroll
    for (int t = 0; t < 4; ++t) s_m[(hg * 4 + t) * HID_ + d] = a4[t];
  }
  __syncthreads();

  // v-projection of aggregated message, then output projection
  if (tid < HID_) {
    const int h = tid >> 4;
    float a = vb[tid];
    for (int k = 0; k < HID_; ++k) a += s_m[h * HID_ + k] * vw[k * HID_ + tid];
    s_o[tid] = a;
  }
  __syncthreads();
  if (tid < HID_) {
    float a = ob[tid];
    for (int k = 0; k < HID_; ++k) a += s_o[k] * ow[k * HID_ + tid];
    msg[(size_t)(b * N_ + i) * HID_ + tid] = a;
  }
}

// ---------------------------------------------------------------------------
extern "C" void kernel_launch(void* const* d_in, const int* in_sizes, int n_in,
                              void* d_out, int out_size, void* d_ws, size_t ws_size,
                              hipStream_t stream) {
  const float* node_h    = (const float*)d_in[0];
  const float* dart_h    = (const float*)d_in[1];
  const float* edge_x    = (const float*)d_in[2];
  const float* edge_mask = (const float*)d_in[3];
  const float* global_x  = (const float*)d_in[4];
  const float* du_w1 = (const float*)d_in[5];
  const float* du_b1 = (const float*)d_in[6];
  const float* du_w2 = (const float*)d_in[7];
  const float* du_b2 = (const float*)d_in[8];
  const float* dn_g  = (const float*)d_in[9];
  const float* dn_b  = (const float*)d_in[10];
  const float* nu_w1 = (const float*)d_in[11];
  const float* nu_b1 = (const float*)d_in[12];
  const float* nu_w2 = (const float*)d_in[13];
  const float* nu_b2 = (const float*)d_in[14];
  const float* nn_g  = (const float*)d_in[15];
  const float* nn_b  = (const float*)d_in[16];
  const float* in_qw  = (const float*)d_in[17];
  const float* in_qb  = (const float*)d_in[18];
  const float* in_kw  = (const float*)d_in[19];
  const float* in_kb  = (const float*)d_in[20];
  const float* in_vw  = (const float*)d_in[21];
  const float* in_vb  = (const float*)d_in[22];
  const float* in_ow  = (const float*)d_in[23];
  const float* in_ob  = (const float*)d_in[24];
  const float* out_qw = (const float*)d_in[25];
  const float* out_qb = (const float*)d_in[26];
  const float* out_kw = (const float*)d_in[27];
  const float* out_kb = (const float*)d_in[28];
  const float* out_vw = (const float*)d_in[29];
  const float* out_vb = (const float*)d_in[30];
  const float* out_ow = (const float*)d_in[31];
  const float* out_ob = (const float*)d_in[32];

  // output layout: node_h | dart_h | in_attn | out_attn
  float* out          = (float*)d_out;
  float* out_node     = out;
  float* out_dart     = out + 131072;                 // B*N*HID
  float* out_in_attn  = out + 131072 + 33554432;      // + B*N*N*HID
  float* out_out_attn = out_in_attn + 2097152;        // + B*H*N*N

  // workspace layout (offsets multiple of 256B)
  char*   ws      = (char*)d_ws;
  bf16_t* w1d     = (bf16_t*)(ws + 0);        // 416*128*2 = 106496
  bf16_t* w2d     = (bf16_t*)(ws + 106496);   // 128*128*2 =  32768
  bf16_t* w1n     = (bf16_t*)(ws + 139264);   // 448*128*2 = 114688
  bf16_t* w2n     = (bf16_t*)(ws + 253952);   // 128*128*2 =  32768
  float*  in_msg  = (float*)(ws + 286720);    // B*N*HID*4 = 524288
  float*  out_msg = (float*)(ws + 286720 + 524288);

  // 1) weights -> bf16, packed into fragment-native [K/32][128][32] layout
  pack_w_kernel<<<(416 * 128 + 255) / 256, 256, 0, stream>>>(du_w1, w1d, 416);
  pack_w_kernel<<<(128 * 128 + 255) / 256, 256, 0, stream>>>(du_w2, w2d, 128);
  pack_w_kernel<<<(448 * 128 + 255) / 256, 256, 0, stream>>>(nu_w1, w1n, 448);
  pack_w_kernel<<<(128 * 128 + 255) / 256, 256, 0, stream>>>(nu_w2, w2n, 128);

  // 2) dart update MLP (WMMA) -> new dart_h directly into d_out
  mlp_ln_kernel<<<(B_ * N_ * N_) / 32, 256, 0, stream>>>(
      node_h, nullptr, dart_h, edge_x, w1d, du_b1, w2d, du_b2,
      dn_g, dn_b, dart_h, out_dart, 416, 0);

  // 3) fused attentions (k/v algebraically folded away)
  dart_attn_kernel<<<B_ * N_, 256, 0, stream>>>(
      node_h, out_dart, edge_mask,
      out_qw, out_qb, out_kw, out_kb, out_vw, out_vb, out_ow, out_ob,
      out_msg, out_out_attn, 0);
  dart_attn_kernel<<<B_ * N_, 256, 0, stream>>>(
      node_h, out_dart, edge_mask,
      in_qw, in_qb, in_kw, in_kb, in_vw, in_vb, in_ow, in_ob,
      in_msg, out_in_attn, 1);

  // 4) node update MLP (WMMA) -> new node_h into d_out
  mlp_ln_kernel<<<(B_ * N_) / 32, 256, 0, stream>>>(
      node_h, in_msg, out_msg, global_x, w1n, nu_b1, w2n, nu_b2,
      nn_g, nn_b, node_h, out_node, 448, 1);
}
